// Decoder_RNN_Gumbel_85985245266064
// MI455X (gfx1250) — compile-verified, matrix operand order checked by reading, not run on hardware
//
#include <hip/hip_runtime.h>
#include <hip/hip_bf16.h>

// ---------------- problem constants ----------------
#define BD   64     // batch
#define TKD  2048   // encoder time
#define TLD  256    // decoder time
#define VD   64     // vocab
#define ED   256    // embed
#define KVD  128    // key/value size
#define HD   512    // lstm hidden
#define MHD  256    // mlp hidden
#define EPSF 1e-10f

typedef __attribute__((ext_vector_type(16))) __bf16 v16bf;
typedef __attribute__((ext_vector_type(8)))  float  v8f;
typedef __attribute__((ext_vector_type(4)))  unsigned int u32x4;
typedef __attribute__((ext_vector_type(8)))  int  i32x8;
typedef __attribute__((ext_vector_type(4)))  int  i32x4;
typedef unsigned short u16;
typedef unsigned int   u32;

// ---------------- bf16 bit helpers ----------------
__device__ inline u16 f2bf(float f) {
    u32 u = __float_as_uint(f);
    u32 lsb = (u >> 16) & 1u;
    u += 0x7fffu + lsb;            // round-to-nearest-even
    return (u16)(u >> 16);
}
__device__ inline float bf2f(u16 b) { return __uint_as_float(((u32)b) << 16); }

union BFrag { v16bf v; u16 s[16]; };

__device__ inline float sigm(float x) { return 1.0f / (1.0f + __expf(-x)); }

// A-fragment (16x32 bf16, MxK) from LDS staged as [64][128] bf16-bits.
__device__ inline v16bf load_a_frag(const u16* As, int row0, int kb, int lane) {
    int m  = row0 + (lane & 15);
    int kh = (lane >> 4) << 3;
    const u16* p = As + m * 128 + kb * 32 + kh;
    BFrag u;
#pragma unroll
    for (int i = 0; i < 8; i++) { u.s[i] = p[i]; u.s[8 + i] = p[16 + i]; }
    return u.v;
}

// ---------------- TDM: async 2-D tile (64 rows x 128 bf16) global -> LDS ----------------
// D# built per CDNA5 ISA ch.8: group0 {count=1, lds_addr, global_addr, type=2},
// group1 {data_size=2B, tensor_dim0=stride, tensor_dim1=64, tile 128x64, dim0_stride}.
__device__ inline void tdm_load_tile(unsigned lds_off, const u16* gsrc, unsigned row_elems) {
#if defined(__gfx1250__)
    unsigned long long ga = (unsigned long long)gsrc;
    u32x4 g0;
    g0[0] = 1u;                                         // count=1, is_restore=0, gather=0
    g0[1] = lds_off;                                    // lds_addr
    g0[2] = (u32)ga;                                    // global_addr[31:0]
    g0[3] = ((u32)(ga >> 32) & 0x01FFFFFFu) | (2u << 30); // addr[56:32] | type=2
    i32x8 g1;
    g1[0] = (int)(1u << 16);                            // workgroup_mask=0, data_size=1 (2B)
    g1[1] = (int)((row_elems & 0xFFFFu) << 16);         // tensor_dim0[15:0] @ bits63:48
    g1[2] = (int)((row_elems >> 16) | (64u << 16));     // tensor_dim0[31:16] | tensor_dim1=64
    g1[3] = (int)(128u << 16);                          // tile_dim0=128 @ bits127:112
    g1[4] = 64;                                         // tile_dim1=64, tile_dim2=0
    g1[5] = (int)row_elems;                             // tensor_dim0_stride[31:0]
    g1[6] = 0;                                          // stride hi | dim1_stride lo
    g1[7] = 0;
    i32x4 z4 = {0, 0, 0, 0};
#if __clang_major__ >= 23
    i32x8 z8 = {0, 0, 0, 0, 0, 0, 0, 0};
    __builtin_amdgcn_tensor_load_to_lds(g0, g1, z4, z4, z8, 0);
#else
    __builtin_amdgcn_tensor_load_to_lds(g0, g1, z4, z4, 0);
#endif
#else
    (void)lds_off; (void)gsrc; (void)row_elems;
#endif
}

__device__ inline void tdm_wait(int pending) {
#if defined(__gfx1250__)
    if (pending) __builtin_amdgcn_s_wait_tensorcnt((short)1);
    else         __builtin_amdgcn_s_wait_tensorcnt((short)0);
#else
    (void)pending;
#endif
}

// ---------------- weight pack: fp32 [N,Ka]+[N,Kb] -> bf16 WMMA B-fragment tiles -------------
__global__ void pack_weights(const float* __restrict__ Wa, int Ka,
                             const float* __restrict__ Wb, int Kb,
                             u16* __restrict__ dst, int N) {
    int K  = Ka + Kb;
    int kT = K >> 5;
    int tile = blockIdx.x * (blockDim.x >> 5) + (threadIdx.x >> 5);
    int lane = threadIdx.x & 31;
    int nTiles = (N >> 4) * kT;
    if (tile >= nTiles) return;
    int n0 = tile / kT, kb = tile % kT;
    int n  = n0 * 16 + (lane & 15);
    int kh = (lane >> 4) << 3;
    u16* d = dst + (size_t)tile * 512 + lane * 16;
#pragma unroll
    for (int i = 0; i < 8; i++) {
        int k0 = kb * 32 + kh + i;
        int k1 = k0 + 16;
        float v0 = (k0 < Ka) ? Wa[(size_t)n * Ka + k0] : Wb[(size_t)n * Kb + (k0 - Ka)];
        float v1 = (k1 < Ka) ? Wa[(size_t)n * Ka + k1] : Wb[(size_t)n * Kb + (k1 - Ka)];
        d[i]     = f2bf(v0);
        d[8 + i] = f2bf(v1);
    }
}

__global__ void add2_kernel(float* d, const float* a, const float* b, int n) {
    int i = blockIdx.x * blockDim.x + threadIdx.x;
    if (i < n) d[i] = a[i] + b[i];
}

__global__ void zero_u32_kernel(u32* p, int n) {
    int i = blockIdx.x * blockDim.x + threadIdx.x;
    if (i < n) p[i] = 0u;
}

// ---------------- gumbel-softmax of labels + SOS row ----------------
__global__ void gumbel_kernel(const int* __restrict__ labels, const int* __restrict__ sos_ptr,
                              float* __restrict__ out_sm,   // [B,TL,V]
                              float* __restrict__ out_lT,   // [B,TL]
                              float* __restrict__ y0) {     // [B,V]
    int w    = blockIdx.x * (blockDim.x >> 5) + (threadIdx.x >> 5);
    int lane = threadIdx.x & 31;
    const int ROWS = (TLD + 1) * BD;
    if (w >= ROWS) return;
    int t = w >> 6, b = w & 63;
    int label; float* dst; u32 salt;
    if (t < TLD) { label = labels[t * BD + b]; dst = out_sm + ((size_t)b * TLD + t) * VD; salt = 0x9E3779B9u; }
    else         { label = *sos_ptr;           dst = y0 + (size_t)b * VD;                salt = 0x85EBCA6Bu; }
    float z[2];
#pragma unroll
    for (int e = 0; e < 2; e++) {
        int v = lane * 2 + e;
        u32 x = salt ^ ((u32)(t * (BD * VD) + b * VD + v) * 0x9E3779B9u);
        x ^= x >> 16; x *= 0x7feb352dU; x ^= x >> 15; x *= 0x846ca68bU; x ^= x >> 16;
        float u = (float)(x >> 8) * (1.0f / 16777216.0f);
        float noise = -__logf(-__logf(u + EPSF) + EPSF);
        z[e] = ((v == label) ? 1.0f : 0.0f) + noise;
    }
    float mx = fmaxf(z[0], z[1]);
#pragma unroll
    for (int o = 16; o > 0; o >>= 1) mx = fmaxf(mx, __shfl_xor(mx, o, 32));
    float p0 = __expf(z[0] - mx), p1 = __expf(z[1] - mx);
    float s = p0 + p1;
#pragma unroll
    for (int o = 16; o > 0; o >>= 1) s += __shfl_xor(s, o, 32);
    float inv = 1.0f / s;
    dst[lane * 2]     = p0 * inv;
    dst[lane * 2 + 1] = p1 * inv;
    if (t < TLD && lane == 0) out_lT[(size_t)b * TLD + t] = (float)label;
}

// ---------------- embedding GEMM (WMMA): xs_emb[t,b,:] = emb(y_{t-1}) ----------------
__global__ __launch_bounds__(256) void embed_kernel(const float* __restrict__ out_sm,
                                                    const float* __restrict__ y0,
                                                    const u16* __restrict__ Wem,
                                                    const float* __restrict__ bemb,
                                                    u16* __restrict__ xe) {  // bf16 [TL,B,E]
    int wid  = blockIdx.x * (blockDim.x >> 5) + (threadIdx.x >> 5);
    int lane = threadIdx.x & 31;
    const int MT = (TLD * BD) >> 4;   // 1024
    if (wid >= MT * (ED >> 4)) return;
    int mt = wid % MT, nt = wid / MT;
    int myrow = mt * 16 + (lane & 15);
    int t = myrow >> 6, b = myrow & 63;
    const float* src = (t == 0) ? (y0 + (size_t)b * VD)
                                : (out_sm + ((size_t)b * TLD + (t - 1)) * VD);
    float bv = bemb[nt * 16 + (lane & 15)];
    v8f acc;
#pragma unroll
    for (int i = 0; i < 8; i++) acc[i] = bv;
#pragma unroll
    for (int kb = 0; kb < VD / 32; kb++) {
        BFrag a;
        int kbase = kb * 32 + ((lane >> 4) << 3);
#pragma unroll
        for (int i = 0; i < 8; i++) {
            a.s[i]     = f2bf(src[kbase + i]);
            a.s[8 + i] = f2bf(src[kbase + 16 + i]);
        }
        v16bf bf = *(const v16bf*)(Wem + ((size_t)(nt * (VD / 32) + kb)) * 512 + lane * 16);
        acc = __builtin_amdgcn_wmma_f32_16x16x32_bf16(false, a.v, false, bf, (short)0, acc, false, false);
    }
#pragma unroll
    for (int v = 0; v < 8; v++) {
        int row = mt * 16 + v + ((lane >> 4) << 3);
        int tt = row >> 6, bb = row & 63;
        int n = nt * 16 + (lane & 15);
        xe[((size_t)tt * BD + bb) * ED + n] = f2bf(acc[v]);
    }
}

// ---------------- fused LSTM step: TDM-staged A, WMMA gate GEMM, activations, state --------
// 16 WGs x 8 waves. WG owns H-cols [wg*32, wg*32+32) across all 4 gates (wave = g*2+j).
// A = [xe | xf | hin], all bf16, staged per 128-col chunk by tensor_load_to_lds, double-buffered.
__global__ __launch_bounds__(256) void lstm_step_kernel(
    const u16* __restrict__ xe, int Ke,
    const u16* __restrict__ xf, int Kf,
    const u16* __restrict__ hin,
    const u16* __restrict__ Wp, const float* __restrict__ bias,
    u16* __restrict__ hout, float* __restrict__ cst, int K) {
    __shared__ u16   As[2][64 * 128];
    __shared__ float Gs[4 * 64 * 32];
    int tid = threadIdx.x;
    int wg  = blockIdx.x;
    int wave = tid >> 5, lane = tid & 31;
    int g = wave >> 1, j = wave & 1;
    int colBase = wg * 32;
    int n0 = g * 32 + wg * 2 + j;   // N-tile index in [0,128)
    int kT = K >> 5;
    int nchunks = K >> 7;
    int Kxf = Ke + Kf;
    float bv = bias[g * HD + colBase + j * 16 + (lane & 15)];
    v8f acc[4];
#pragma unroll
    for (int m = 0; m < 4; m++) {
#pragma unroll
        for (int i = 0; i < 8; i++) acc[m][i] = bv;
    }
    // chunk -> (source base, row stride): chunks never straddle sources (Ke,Kf % 128 == 0)
    auto chunk_src = [&](int cb, const u16** src, unsigned* stride) {
        if (cb < Ke)       { *src = xe  + cb;         *stride = (unsigned)Ke; }
        else if (cb < Kxf) { *src = xf  + (cb - Ke);  *stride = (unsigned)Kf; }
        else               { *src = hin + (cb - Kxf); *stride = (unsigned)HD; }
    };
    if (tid < 32) {   // wave 0 drives the TDM pipeline
        const u16* s0; unsigned st0;
        chunk_src(0, &s0, &st0);
        tdm_load_tile((unsigned)(unsigned long long)(void*)&As[0][0], s0, st0);
    }
    for (int ch = 0; ch < nchunks; ch++) {
        if (tid < 32) {
            if (ch + 1 < nchunks) {
                const u16* sn; unsigned stn;
                chunk_src((ch + 1) << 7, &sn, &stn);
                tdm_load_tile((unsigned)(unsigned long long)(void*)&As[(ch + 1) & 1][0], sn, stn);
                tdm_wait(1);   // chunk ch complete (TDM in-order per wave)
            } else {
                tdm_wait(0);
            }
        }
        __syncthreads();
        const u16* Ab = &As[ch & 1][0];
        __builtin_prefetch(Wp + ((size_t)n0 * kT + (size_t)(ch + 1) * 4) * 512, 0, 1);
#pragma unroll
        for (int kb = 0; kb < 4; kb++) {
            v16bf bf = *(const v16bf*)(Wp + ((size_t)n0 * kT + (size_t)(ch * 4 + kb)) * 512 + lane * 16);
#pragma unroll
            for (int m = 0; m < 4; m++) {
                v16bf a = load_a_frag(Ab, m * 16, kb, lane);
                acc[m] = __builtin_amdgcn_wmma_f32_16x16x32_bf16(false, a, false, bf, (short)0, acc[m], false, false);
            }
        }
        __syncthreads();   // all waves done with buf before TDM overwrites it next iter
    }
    // gates -> LDS
#pragma unroll
    for (int m = 0; m < 4; m++) {
#pragma unroll
        for (int v = 0; v < 8; v++) {
            int row = m * 16 + v + ((lane >> 4) << 3);
            int col = j * 16 + (lane & 15);
            Gs[(g * 64 + row) * 32 + col] = acc[m][v];
        }
    }
    __syncthreads();
    // elementwise LSTM update on this WG's 32-col slice (i,f,g,o gate order)
    for (int idx = tid; idx < 64 * 32; idx += 256) {
        int row = idx >> 5, col = idx & 31;
        int gc = colBase + col;
        float ig = sigm(Gs[(0 * 64 + row) * 32 + col]);
        float fg = sigm(Gs[(1 * 64 + row) * 32 + col]);
        float gg = tanhf(Gs[(2 * 64 + row) * 32 + col]);
        float og = sigm(Gs[(3 * 64 + row) * 32 + col]);
        float c0 = cst[(size_t)row * HD + gc];
        float c2 = fg * c0 + ig * gg;
        cst[(size_t)row * HD + gc]  = c2;
        hout[(size_t)row * HD + gc] = f2bf(og * tanhf(c2));
    }
}

// ---------------- generic small GEMM (M=64, bf16 A): out = act([A1|A2] @ Wp^T + bias) ------
__global__ __launch_bounds__(256) void gemm64_kernel(
    const u16* __restrict__ A1, int K1, const u16* __restrict__ A2, int K2,
    const u16* __restrict__ Wp, const float* __restrict__ bias,
    void* __restrict__ outp, long ldo, int Ntiles, int act, int obf) {
    int wid  = blockIdx.x * (blockDim.x >> 5) + (threadIdx.x >> 5);
    int lane = threadIdx.x & 31;
    if (wid >= 4 * Ntiles) return;
    int mt = wid & 3, nt = wid >> 2;
    int K = K1 + K2, kT = K >> 5;
    int m = mt * 16 + (lane & 15);
    float bv = bias[nt * 16 + (lane & 15)];
    v8f acc;
#pragma unroll
    for (int i = 0; i < 8; i++) acc[i] = bv;
    for (int kb = 0; kb < kT; kb++) {
        int c0 = kb * 32;     // whole 32-col chunk lies in one source (K1 % 32 == 0)
        const u16* srow = (c0 < K1) ? (A1 + (size_t)m * K1 + c0)
                                    : (A2 + (size_t)m * K2 + (c0 - K1));
        const u16* p = srow + ((lane >> 4) << 3);
        BFrag a;
#pragma unroll
        for (int i = 0; i < 8; i++) { a.s[i] = p[i]; a.s[8 + i] = p[16 + i]; }
        v16bf bf = *(const v16bf*)(Wp + ((size_t)nt * kT + kb) * 512 + lane * 16);
        acc = __builtin_amdgcn_wmma_f32_16x16x32_bf16(false, a.v, false, bf, (short)0, acc, false, false);
    }
#pragma unroll
    for (int v = 0; v < 8; v++) {
        float x = acc[v];
        if (act == 1) x = (x >= 0.0f) ? x : 0.9f * x;   // LeakyReLU(0.9)
        int row = mt * 16 + v + ((lane >> 4) << 3);
        int n   = nt * 16 + (lane & 15);
        if (obf) ((u16*)outp)[(size_t)row * ldo + n] = f2bf(x);
        else     ((float*)outp)[(size_t)row * ldo + n] = x;
    }
}

// ---------------- attention: energy + mask-mul softmax + context, one WG per batch ----------
__global__ __launch_bounds__(256) void attention_kernel(
    const float* __restrict__ enc_key, const float* __restrict__ val,
    const int* __restrict__ lens, const u16* __restrict__ query,
    u16* __restrict__ ctx_out, float* __restrict__ attn_base, int t) {
    __shared__ float q[KVD];
    __shared__ float e[TKD];
    __shared__ float red[8];
    __shared__ float cpart[2][KVD];
    int b = blockIdx.x, tid = threadIdx.x;
    int wv = tid >> 5, lane = tid & 31;
    if (tid < KVD) q[tid] = bf2f(query[(size_t)b * KVD + tid]);
    __syncthreads();
    int len = lens[b];
    const float* ek = enc_key + (size_t)b * TKD * KVD;
    float lmax = -3.4e38f;
#pragma unroll
    for (int r = 0; r < 8; r++) {
        int tk = tid + r * 256;
        const float4* kp = (const float4*)(ek + (size_t)tk * KVD);
        const float4* qp = (const float4*)q;
        float s = 0.0f;
#pragma unroll
        for (int i = 0; i < KVD / 4; i++) {
            float4 k4 = kp[i], q4 = qp[i];
            s += k4.x * q4.x + k4.y * q4.y + k4.z * q4.z + k4.w * q4.w;
        }
        float msk = (tk < len) ? s : 0.0f;   // energy * mask (mask-by-multiply)
        e[tk] = msk;
        lmax = fmaxf(lmax, msk);
    }
#pragma unroll
    for (int o = 16; o > 0; o >>= 1) lmax = fmaxf(lmax, __shfl_xor(lmax, o, 32));
    if (lane == 0) red[wv] = lmax;
    __syncthreads();
    float mx = red[0];
#pragma unroll
    for (int i = 1; i < 8; i++) mx = fmaxf(mx, red[i]);
    __syncthreads();
    float lsum = 0.0f;
#pragma unroll
    for (int r = 0; r < 8; r++) {
        int tk = tid + r * 256;
        float p = __expf(e[tk] - mx);
        e[tk] = p;
        lsum += p;
    }
#pragma unroll
    for (int o = 16; o > 0; o >>= 1) lsum += __shfl_xor(lsum, o, 32);
    if (lane == 0) red[wv] = lsum;
    __syncthreads();
    float s = 0.0f;
#pragma unroll
    for (int i = 0; i < 8; i++) s += red[i];
    float inv = 1.0f / s;
#pragma unroll
    for (int r = 0; r < 8; r++) {
        int tk = tid + r * 256;
        float a = e[tk] * inv;
        e[tk] = a;
        attn_base[(size_t)b * TKD * TLD + (size_t)tk * TLD + t] = a;
    }
    __syncthreads();
    int half = tid >> 7, k = tid & 127;
    const float* vb = val + (size_t)b * TKD * KVD;
    float accv = 0.0f;
    for (int tk = half * 1024; tk < (half + 1) * 1024; tk++)
        accv += e[tk] * vb[(size_t)tk * KVD + k];
    cpart[half][k] = accv;
    __syncthreads();
    if (tid < KVD) ctx_out[(size_t)b * KVD + tid] = f2bf(cpart[0][tid] + cpart[1][tid]);
}

// ---------------- argmax over V per (b,t) ----------------
__global__ void argmax_kernel(const float* __restrict__ yhat, float* __restrict__ outlab) {
    int idx = blockIdx.x * blockDim.x + threadIdx.x;
    if (idx >= BD * TLD) return;
    const float* r = yhat + (size_t)idx * VD;
    float best = r[0]; int bi = 0;
#pragma unroll 4
    for (int v = 1; v < VD; v++) { float x = r[v]; if (x > best) { best = x; bi = v; } }
    outlab[idx] = (float)bi;
}

// =====================================================================================
extern "C" void kernel_launch(void* const* d_in, const int* in_sizes, int n_in,
                              void* d_out, int out_size, void* d_ws, size_t ws_size,
                              hipStream_t stream) {
    (void)in_sizes; (void)n_in; (void)out_size; (void)ws_size;
    const float* enc_key = (const float*)d_in[0];
    const float* value   = (const float*)d_in[1];
    const int*   labels  = (const int*)d_in[2];
    const int*   lens    = (const int*)d_in[3];
    const int*   sos     = (const int*)d_in[4];
    const float* w_emb = (const float*)d_in[6],  *b_emb = (const float*)d_in[7];
    const float* w_ih0 = (const float*)d_in[8],  *w_hh0 = (const float*)d_in[9];
    const float* b_ih0 = (const float*)d_in[10], *b_hh0 = (const float*)d_in[11];
    const float* w_ih1 = (const float*)d_in[12], *w_hh1 = (const float*)d_in[13];
    const float* b_ih1 = (const float*)d_in[14], *b_hh1 = (const float*)d_in[15];
    const float* w_ih2 = (const float*)d_in[16], *w_hh2 = (const float*)d_in[17];
    const float* b_ih2 = (const float*)d_in[18], *b_hh2 = (const float*)d_in[19];
    const float* w_fc  = (const float*)d_in[20], *b_fc  = (const float*)d_in[21];
    const float* w_mlp1= (const float*)d_in[22], *b_mlp1= (const float*)d_in[23];
    const float* w_mlp2= (const float*)d_in[24], *b_mlp2= (const float*)d_in[25];

    // ---- workspace carve-up (~22 MB) ----
    char* wp = (char*)d_ws;
    auto alloc = [&](size_t bytes) -> char* {
        char* p = wp; wp += (bytes + 255) & ~(size_t)255; return p;
    };
    u16* W0p = (u16*)alloc((size_t)2048 * 896  * 2);  // [x(384)|h(512)] combined
    u16* W1p = (u16*)alloc((size_t)2048 * 1024 * 2);
    u16* W2p = (u16*)alloc((size_t)2048 * 1024 * 2);
    u16* Wfc = (u16*)alloc((size_t)128  * 512  * 2);
    u16* Wm1 = (u16*)alloc((size_t)256  * 256  * 2);
    u16* Wm2 = (u16*)alloc((size_t)64   * 256  * 2);
    u16* Wem = (u16*)alloc((size_t)256  * 64   * 2);
    float* bcomb = (float*)alloc((size_t)3 * 2048 * 4);
    float* y0    = (float*)alloc((size_t)BD * VD * 4);
    u16*   xsemb = (u16*)alloc((size_t)TLD * BD * ED * 2);
    // contiguous zero-init state block: c (fp32) + h (bf16) + ctx (bf16)
    float* cstate   = (float*)alloc((size_t)3 * BD * HD * 4);       // 393216 B
    u16*   hstate   = (u16*)alloc((size_t)3 * 2 * BD * HD * 2);     // 393216 B
    u16*   ctxstate = (u16*)alloc((size_t)2 * BD * KVD * 2);        //  32768 B
    u16*   qbuf = (u16*)alloc((size_t)BD * KVD * 2);
    u16*   mlph = (u16*)alloc((size_t)BD * MHD * 2);

    float* out    = (float*)d_out;
    float* o_yhat = out;                 // [B,TL,V]  1048576
    float* o_lab  = out + 1048576;       // [B,TL]      16384
    float* o_lT   = out + 1064960;       // [B,TL]      16384
    float* o_sm   = out + 1081344;       // [B,TL,V]  1048576
    float* o_attn = out + 2129920;       // [B,TK,TL] 33554432

    auto cdiv = [](int a, int b) { return (a + b - 1) / b; };

    // ---- one-time (per call) prep ----
    int tiles;
    tiles = (2048/16)*((384+512)/32);
    pack_weights<<<cdiv(tiles,8),256,0,stream>>>(w_ih0,384,w_hh0,512,W0p,2048);
    tiles = (2048/16)*((512+512)/32);
    pack_weights<<<cdiv(tiles,8),256,0,stream>>>(w_ih1,512,w_hh1,512,W1p,2048);
    pack_weights<<<cdiv(tiles,8),256,0,stream>>>(w_ih2,512,w_hh2,512,W2p,2048);
    tiles = (128/16)*(512/32);
    pack_weights<<<cdiv(tiles,8),256,0,stream>>>(w_fc,512,nullptr,0,Wfc,128);
    tiles = (256/16)*(256/32);
    pack_weights<<<cdiv(tiles,8),256,0,stream>>>(w_mlp1,256,nullptr,0,Wm1,256);
    tiles = (64/16)*(256/32);
    pack_weights<<<cdiv(tiles,8),256,0,stream>>>(w_mlp2,256,nullptr,0,Wm2,64);
    tiles = (256/16)*(64/32);
    pack_weights<<<cdiv(tiles,8),256,0,stream>>>(w_emb,64,nullptr,0,Wem,256);

    add2_kernel<<<8,256,0,stream>>>(bcomb,        b_ih0, b_hh0, 2048);
    add2_kernel<<<8,256,0,stream>>>(bcomb + 2048, b_ih1, b_hh1, 2048);
    add2_kernel<<<8,256,0,stream>>>(bcomb + 4096, b_ih2, b_hh2, 2048);
    {   // zero the contiguous state block (c + h + ctx), all 256B-aligned and adjacent
        int zeros = (393216 + 393216 + 32768) / 4;
        zero_u32_kernel<<<cdiv(zeros,256),256,0,stream>>>((u32*)cstate, zeros);
    }

    gumbel_kernel<<<cdiv((TLD+1)*BD,8),256,0,stream>>>(labels, sos, o_sm, o_lT, y0);
    embed_kernel<<<cdiv(1024*16,8),256,0,stream>>>(o_sm, y0, Wem, b_emb, xsemb);

    u16* hb[3][2]; float* cb[3]; u16* cx[2];
    for (int L = 0; L < 3; L++) {
        for (int p = 0; p < 2; p++) hb[L][p] = hstate + (size_t)(L*2 + p) * BD * HD;
        cb[L] = cstate + (size_t)L * BD * HD;
    }
    cx[0] = ctxstate;
    cx[1] = ctxstate + (size_t)BD * KVD;

    // ---- sequential decode: 256 steps, ping-pong state ----
    for (int t = 0; t < TLD; t++) {
        int p = t & 1, q = 1 - p;
        lstm_step_kernel<<<16,256,0,stream>>>(xsemb + (size_t)t*BD*ED, 256,
                                              cx[p], 128, hb[0][p],
                                              W0p, bcomb, hb[0][q], cb[0], 896);
        lstm_step_kernel<<<16,256,0,stream>>>(nullptr, 0,
                                              hb[0][q], 512, hb[1][p],
                                              W1p, bcomb + 2048, hb[1][q], cb[1], 1024);
        lstm_step_kernel<<<16,256,0,stream>>>(nullptr, 0,
                                              hb[1][q], 512, hb[2][p],
                                              W2p, bcomb + 4096, hb[2][q], cb[2], 1024);
        gemm64_kernel<<<4,256,0,stream>>>(hb[2][q], 512, nullptr, 0,
                                          Wfc, b_fc, qbuf, 128, 8, 0, 1);
        attention_kernel<<<64,256,0,stream>>>(enc_key, value, lens, qbuf,
                                              cx[q], o_attn, t);
        gemm64_kernel<<<8,256,0,stream>>>(qbuf, 128, cx[q], 128,
                                          Wm1, b_mlp1, mlph, 256, 16, 1, 1);
        gemm64_kernel<<<2,256,0,stream>>>(mlph, 256, nullptr, 0,
                                          Wm2, b_mlp2, o_yhat + (size_t)t * VD,
                                          (long)TLD * VD, 4, 0, 0);
    }
    argmax_kernel<<<64,256,0,stream>>>(o_yhat, o_lab);
}
